// RankOnePlanes_89240830476840
// MI455X (gfx1250) — compile-verified, matrix operand order checked by reading, not run on hardware
//
#include <hip/hip_runtime.h>
#include <hip/hip_bf16.h>

typedef __attribute__((ext_vector_type(16))) _Float16 v16h;
typedef __attribute__((ext_vector_type(8)))  float    v8f;
typedef __attribute__((ext_vector_type(4)))  float    f4;

#define TLEN      128          // 1D table length
#define CL        32           // channels per line
#define HID       128          // hidden width
#define LDS_PITCH 36           // padded row pitch (floats) to spread LDS banks
#define INV_R     2.5f         // 1/RANGE
#define INV_R2    6.25f        // 1/RANGE^2

__device__ __forceinline__ void load8(const float* p, float g[8]) {
    const f4 lo = *reinterpret_cast<const f4*>(p);
    const f4 hi = *reinterpret_cast<const f4*>(p + 4);
    g[0] = lo[0]; g[1] = lo[1]; g[2] = lo[2]; g[3] = lo[3];
    g[4] = hi[0]; g[5] = hi[1]; g[6] = hi[2]; g[7] = hi[3];
}

__global__ __launch_bounds__(256)
void rank1planes_wmma_kernel(const float* __restrict__ coords,
                             const float* __restrict__ lines,
                             const float* __restrict__ W1,
                             const float* __restrict__ b1,
                             const float* __restrict__ W2,
                             const float* __restrict__ b2,
                             float* __restrict__ out,
                             int nTiles) {
    // ---- stage lines (3,128,32) into LDS with padded pitch (bank spread) ----
    __shared__ float s_lines[3 * TLEN * LDS_PITCH];     // 55,296 B
    {
        const int nt = blockDim.x;
        for (int i = threadIdx.x; i < 3 * TLEN * (CL / 4); i += nt) {
            const int a  = i >> 10;          // / (128*8)
            const int r  = (i >> 3) & 127;
            const int c4 = i & 7;
            *reinterpret_cast<f4*>(&s_lines[a * (TLEN * LDS_PITCH) + r * LDS_PITCH + c4 * 4]) =
                reinterpret_cast<const f4*>(lines)[i];
        }
    }
    __syncthreads();

    const int lane  = threadIdx.x & 31;
    const int ln    = lane & 15;             // row/col within 16
    const int lhalf = lane >> 4;             // 0: K 0..7/16..23, 1: K 8..15/24..31
    const int kbase = lhalf * 8;

    // ---- preload W1 as 8 B-tiles (f16) in registers: lane ln holds h = 16t+ln ----
    v16h Bt[8];
    float b1v[8], w2v[8];
    #pragma unroll
    for (int t = 0; t < 8; ++t) {
        const float* wr = W1 + (t * 16 + ln) * CL + kbase;
        float g0[8], g1[8];
        load8(wr,      g0);                  // K = kbase..kbase+7
        load8(wr + 16, g1);                  // K = kbase+16..kbase+23
        #pragma unroll
        for (int e = 0; e < 8; ++e) {
            Bt[t][e]     = (_Float16)g0[e];
            Bt[t][8 + e] = (_Float16)g1[e];
        }
        b1v[t] = b1[t * 16 + ln];
        w2v[t] = W2[t * 16 + ln];
    }
    const float b2s = b2[0];

    const int wavesPerBlock = blockDim.x >> 5;
    const int waveId     = blockIdx.x * wavesPerBlock + (threadIdx.x >> 5);
    const int waveStride = gridDim.x * wavesPerBlock;

    for (int tile = waveId; tile < nTiles; tile += waveStride) {
        const int m = tile * 16 + ln;                 // this lane's point
        const float* cp = coords + (size_t)m * 3;
        const float cx = cp[0], cy = cp[1], cz = cp[2];

        if (tile + waveStride < nTiles) {             // wave-uniform -> global_prefetch
            __builtin_prefetch(coords + (size_t)(tile + waveStride) * 48 + (size_t)ln * 3, 0, 0);
        }

        // ---- per-axis interpolation params (align_corners=True, zeros padding) ----
        float w0[3], w1[3];
        int   r0[3], r1[3];
        const float c3[3] = {cx, cy, cz};
        #pragma unroll
        for (int a = 0; a < 3; ++a) {
            const float pos = (c3[a] + 1.0f) * (0.5f * (float)(TLEN - 1));
            const float fl  = floorf(pos);
            const float fr  = pos - fl;
            const int i0 = (int)fl;
            const int i1 = i0 + 1;
            w0[a] = (i0 >= 0 && i0 < TLEN) ? (1.0f - fr) : 0.0f;
            w1[a] = (i1 >= 0 && i1 < TLEN) ? fr : 0.0f;
            const int c0 = min(max(i0, 0), TLEN - 1);
            const int c1 = min(max(i1, 0), TLEN - 1);
            r0[a] = a * (TLEN * LDS_PITCH) + c0 * LDS_PITCH;
            r1[a] = a * (TLEN * LDS_PITCH) + c1 * LDS_PITCH;
        }

        // ---- build A-matrix row fragment: 16 of 32 channels for this lane ----
        v16h A;
        #pragma unroll
        for (int ch = 0; ch < 2; ++ch) {
            const int off = kbase + ch * 16;
            float gx0[8], gx1[8], gy0[8], gy1[8], gz0[8], gz1[8];
            load8(&s_lines[r0[0] + off], gx0);  load8(&s_lines[r1[0] + off], gx1);
            load8(&s_lines[r0[1] + off], gy0);  load8(&s_lines[r1[1] + off], gy1);
            load8(&s_lines[r0[2] + off], gz0);  load8(&s_lines[r1[2] + off], gz1);
            #pragma unroll
            for (int j = 0; j < 8; ++j) {
                const float xe = gx0[j] * w0[0] + gx1[j] * w1[0];
                const float ye = gy0[j] * w0[1] + gy1[j] * w1[1];
                const float ze = gz0[j] * w0[2] + gz1[j] * w1[2];
                const float f1 = xe + ye + ze;
                const float f2 = (xe * ye + (xe + ye) * ze) * INV_R;
                const float f3 = (xe * ye) * ze * INV_R2;
                A[ch * 8 + j] = (_Float16)(f1 + f2 + f3);
            }
        }

        // ---- layer 1 on WMMA, fused relu + layer-2 dot ----
        float partial[8] = {0.f, 0.f, 0.f, 0.f, 0.f, 0.f, 0.f, 0.f};
        #pragma unroll
        for (int t = 0; t < 8; ++t) {
            v8f acc = {};
            acc = __builtin_amdgcn_wmma_f32_16x16x32_f16(
                false, A, false, Bt[t], (short)0, acc, false, false);
            #pragma unroll
            for (int r = 0; r < 8; ++r) {
                float h = acc[r] + b1v[t];          // row M = r + 8*lhalf, N = ln + 16t
                h = h > 0.0f ? h : 0.0f;
                partial[r] += h * w2v[t];
            }
        }

        // ---- reduce over N (16 lanes within each half-wave) ----
        #pragma unroll
        for (int r = 0; r < 8; ++r) {
            float v = partial[r];
            v += __shfl_xor(v, 1, 32);
            v += __shfl_xor(v, 2, 32);
            v += __shfl_xor(v, 4, 32);
            v += __shfl_xor(v, 8, 32);
            partial[r] = v;
        }

        if (ln == 0) {
            float* o = out + tile * 16 + lhalf * 8;  // lane0: rows 0-7, lane16: rows 8-15
            #pragma unroll
            for (int r = 0; r < 8; ++r) o[r] = partial[r] + b2s;
        }
    }
}

extern "C" void kernel_launch(void* const* d_in, const int* in_sizes, int n_in,
                              void* d_out, int out_size, void* d_ws, size_t ws_size,
                              hipStream_t stream) {
    const float* coords = (const float*)d_in[0];
    const float* lines  = (const float*)d_in[1];
    const float* W1     = (const float*)d_in[2];
    const float* b1     = (const float*)d_in[3];
    const float* W2     = (const float*)d_in[4];
    const float* b2     = (const float*)d_in[5];
    float* out = (float*)d_out;

    const int Mpts   = in_sizes[0] / 3;      // coords is (1, M, 3)
    const int nTiles = Mpts / 16;            // 16 points per wave-tile

    int blocks = (nTiles + 7) / 8;           // ~1 tile per wave at 8 waves/block
    if (blocks > 1024) blocks = 1024;
    if (blocks < 1)    blocks = 1;

    rank1planes_wmma_kernel<<<blocks, 256, 0, stream>>>(
        coords, lines, W1, b1, W2, b2, out, nTiles);
}
